// ViscoplasticMaterialModel_91216515432955
// MI455X (gfx1250) — compile-verified, matrix operand order checked by reading, not run on hardware
//
#include <hip/hip_runtime.h>
#include <hip/hip_bf16.h>

typedef __attribute__((ext_vector_type(16))) _Float16 v16h;
typedef __attribute__((ext_vector_type(8)))  _Float16 v8h;
typedef __attribute__((ext_vector_type(8)))  float    v8f;

#if __has_builtin(__builtin_amdgcn_tensor_load_to_lds) && __has_builtin(__builtin_amdgcn_s_wait_tensorcnt)
#define VP_HAVE_TDM 1
typedef __attribute__((ext_vector_type(4))) unsigned int u32x4_t;
typedef __attribute__((ext_vector_type(8))) int i32x8_t;
typedef __attribute__((ext_vector_type(4))) int i32x4_t;
#else
#define VP_HAVE_TDM 0
#endif

namespace vp {

constexpr int BATCH  = 4096;
constexpr int TSTEPS = 128;
constexpr int SEQL   = 128;
constexpr int WCH    = 32;   // FNO width
constexpr int WFD    = 64;   // width_final
constexpr int ODIM   = 32;   // out_dim == XD
constexpr int ZDIM   = 64;   // == UD
constexpr int NMODES = 16;
constexpr int NLAY   = 3;
constexpr float DTC  = 0.01f;

// ---- d_in indices: jax tree_flatten -> dict keys sorted alphabetically ----
enum {
  IN_E = 0, IN_Y, IN_e, IN_edot, IN_n,
  IN_Wu, IN_Wuz0, IN_Wuz1, IN_Wuz2, IN_Wy0, IN_Wy1, IN_Wy2,
  IN_Wyu0, IN_Wyu1, IN_Wyu2, IN_Wz1, IN_Wz2, IN_Wzu1, IN_Wzu2,
  IN_bu, IN_byu0, IN_byu1, IN_byu2, IN_bz0, IN_bz1, IN_bz2, IN_bzu1, IN_bzu2,
  IN_fc0_b, IN_fc0_w, IN_fc1_b, IN_fc1_w, IN_fc2_b, IN_fc2_w,
  IN_pw0_b, IN_pw0_w, IN_pw1_b, IN_pw1_w, IN_pw2_b, IN_pw2_w,
  IN_sw0_i, IN_sw0_r, IN_sw1_i, IN_sw1_r, IN_sw2_i, IN_sw2_r
};

// ---- workspace layout (bytes); all regions >=2KB-aligned by construction ----
constexpr size_t WS_M1   = 0;                                   // B f32
constexpr size_t WS_MF2  = WS_M1  + (size_t)BATCH*4;            // B*32 f32
constexpr size_t WS_C0   = WS_MF2 + (size_t)BATCH*ODIM*4;       // B*64 f32
constexpr size_t WS_C1   = WS_C0  + (size_t)BATCH*ZDIM*4;
constexpr size_t WS_G1   = WS_C1  + (size_t)BATCH*ZDIM*4;
constexpr size_t WS_G2   = WS_G1  + (size_t)BATCH*ZDIM*4;
constexpr size_t WS_A0   = WS_G2  + (size_t)BATCH*ZDIM*4;       // B f32
constexpr size_t WS_Y1   = WS_A0  + (size_t)BATCH*4;
constexpr size_t WS_Y2   = WS_Y1  + (size_t)BATCH*4;
constexpr size_t WS_SWRP = WS_Y2  + (size_t)BATCH*4;            // packed B tiles
constexpr size_t WS_SWIP = WS_SWRP + (size_t)NLAY*NMODES*2*512*2;
constexpr size_t WS_PW   = WS_SWIP + (size_t)NLAY*NMODES*2*512*2;  // [o][i] (A role)
constexpr size_t WS_FC1P = WS_PW  + (size_t)NLAY*WCH*WCH*2;     // 4 packed tiles
constexpr size_t WS_FC2P = WS_FC1P + (size_t)4*512*2;           // 4 packed tiles
constexpr size_t WS_AWZ1P= WS_FC2P + (size_t)4*512*2;           // 8 packed tiles

// ------------- WMMA helpers (wave32, v_wmma_f32_16x16x32_f16) -------------
__device__ __forceinline__ int lane_id() { return (int)(threadIdx.x & 31); }

// packed-B WMMA lane layout: K index for (lane, j)
__device__ __forceinline__ int kof(int lane, int j) {
  int kb = (lane < 16) ? 0 : 8;
  return (j < 8) ? (kb + j) : (16 + kb + (j - 8));
}

// A fragment 16x32 (MxK), K contiguous in memory -> 2x b128 per lane
__device__ __forceinline__ v16h frag_a_c(const _Float16* p, int ms) {
  int l = lane_id(); int m = l & 15; int kb = (l < 16) ? 0 : 8;
  v8h lo = *(const v8h*)(p + m*ms + kb);
  v8h hi = *(const v8h*)(p + m*ms + 16 + kb);
  return __builtin_shufflevector(lo, hi, 0,1,2,3,4,5,6,7,8,9,10,11,12,13,14,15);
}
// B fragment from pre-swizzled tile (512 halves): lane's 16 halves contiguous
__device__ __forceinline__ v16h frag_b_p(const _Float16* p) {
  return *(const v16h*)(p + (lane_id() << 4));
}
// B fragment generic gather (dynamic operands only)
__device__ __forceinline__ v16h frag_b_g(const _Float16* p, int ks, int ns) {
  int l = lane_id(); int n = l & 15; int kb = (l < 16) ? 0 : 8;
  v16h b;
#pragma unroll
  for (int j = 0; j < 8; ++j) {
    b[j]     = p[(kb + j)*ks + n*ns];
    b[8 + j] = p[(16 + kb + j)*ks + n*ns];
  }
  return b;
}
// D store, M contiguous in memory (stride 1), N stride ns -> 1x b128 per lane
__device__ __forceinline__ void frag_store_h_mc(_Float16* p, int ns, v8f d) {
  int l = lane_id(); int n = l & 15, mb = (l < 16) ? 0 : 8;
  v8h h;
#pragma unroll
  for (int r = 0; r < 8; ++r) h[r] = (_Float16)d[r];
  *(v8h*)(p + n*ns + mb) = h;
}
// D store generic strided (f16)
__device__ __forceinline__ void frag_store_h_g(_Float16* p, int ms, int ns, v8f d) {
  int l = lane_id(); int n = l & 15, mb = (l < 16) ? 0 : 8;
#pragma unroll
  for (int r = 0; r < 8; ++r) p[(mb + r)*ms + n*ns] = (_Float16)d[r];
}
// D store generic strided (f32)
__device__ __forceinline__ void frag_store_f(float* p, int ms, int ns, v8f d) {
  int l = lane_id(); int n = l & 15, mb = (l < 16) ? 0 : 8;
#pragma unroll
  for (int r = 0; r < 8; ++r) p[(mb + r)*ms + n*ns] = d[r];
}
__device__ __forceinline__ v8f wmma_f16(v16h a, v16h b, v8f c) {
  return __builtin_amdgcn_wmma_f32_16x16x32_f16(false, a, false, b, (short)0, c, false, false);
}

__device__ __forceinline__ float gelu_tanh(float x) {
  const float k0 = 0.7978845608028654f, k1 = 0.044715f;
  return 0.5f * x * (1.0f + tanhf(k0 * (x + k1 * x * x * x)));
}

// ==================== kernel 0: m1 + weight staging/swizzling ====================
struct PrepArgs {
  const float* E;
  const float* swr[NLAY];
  const float* swi[NLAY];
  const float* pww[NLAY];
  const float* fc1w;
  const float* fc2w;
  const float* Wz1;
  float* m1;
  _Float16 *swrp, *swip, *pw16, *fc1p, *fc2p, *awz1p;
};

__global__ void __launch_bounds__(256) prep_kernel(PrepArgs a) {
  int g = blockIdx.x * blockDim.x + threadIdx.x;
  if (g < BATCH) {                      // harmonic mean of modulus field
    float s = 0.f;
    for (int l = 0; l < SEQL; ++l) s += 1.0f / a.E[g*SEQL + l];
    a.m1[g] = (float)SEQL / s;
    return;
  }
  g -= BATCH;
  constexpr int NSW = NLAY * NMODES * 2 * 512;     // packed spectral tiles
  if (g < NSW) {
    int j = g & 15, ln = (g >> 4) & 31, nt = (g >> 9) & 1, md = (g >> 10) & 15, l = g >> 14;
    int i = kof(ln, j), o = nt*16 + (ln & 15);
    int src = (i*WCH + o)*NMODES + md;             // params layout [i][o][m]
    a.swrp[g] = (_Float16)a.swr[l][src];
    a.swip[g] = (_Float16)a.swi[l][src];
    return;
  }
  g -= NSW;
  if (g < NLAY * WCH * WCH) {           // pointwise [o][i] copy (A-role operand)
    int l = g / (WCH*WCH);
    a.pw16[g] = (_Float16)a.pww[l][g - l*WCH*WCH];
    return;
  }
  g -= NLAY * WCH * WCH;
  if (g < 4*512) {                      // fc1 packed: tiles nt=0..3, K=c, N=h
    int j = g & 15, ln = (g >> 4) & 31, nt = g >> 9;
    int c = kof(ln, j), h = nt*16 + (ln & 15);
    a.fc1p[g] = (_Float16)a.fc1w[h*WCH + c];
    return;
  }
  g -= 4*512;
  if (g < 4*512) {                      // fc2 packed: tiles (ks,nt), K=h, N=o
    int j = g & 15, ln = (g >> 4) & 31, nt = (g >> 9) & 1, ks = g >> 10;
    int h = ks*32 + kof(ln, j), o = nt*16 + (ln & 15);
    a.fc2p[g] = (_Float16)a.fc2w[o*WFD + h];
    return;
  }
  g -= 4*512;
  if (g < 8*512) {                      // |Wz1|^T packed: tiles (ks,nt), K=j, N=z
    int j = g & 15, ln = (g >> 4) & 31, nt = (g >> 9) & 3, ks = g >> 11;
    int jj = ks*32 + kof(ln, j), z = nt*16 + (ln & 15);
    a.awz1p[g] = (_Float16)fabsf(a.Wz1[z*ZDIM + jj]);
  }
}

// ==================== kernel 1: fused FNO encoder (WMMA) ====================
struct FnoArgs {
  const float *Y, *nn, *ed;
  const float *fc0w, *fc0b, *fc1b, *fc2b;
  const float* pwb[NLAY];
  const _Float16 *swrp, *swip, *pw16, *fc1p, *fc2p;
  float* mf2;
};

constexpr size_t FNO_LDS_HALVES =
    (size_t)16*WCH*SEQL       // x      [16*32][128]   (row-major in l)
  + (size_t)512*32            // xfT    [32 col][512 row] (col-major spectrum)
  + (size_t)512*32            // of     [512][32]      (row-major)
  + (size_t)8*512             // Cfp    8 packed B tiles
  + (size_t)8*512;            // Cip    8 packed B tiles
constexpr size_t FNO_LDS_BYTES = FNO_LDS_HALVES * 2;

__global__ void __launch_bounds__(256) fno_kernel(FnoArgs a) {
  extern __shared__ _Float16 smem[];
  _Float16* xT  = smem;                 // [bb*32+c][l]
  _Float16* xfT = xT + 16*WCH*SEQL;     // [col][bb*32+c]
  _Float16* of  = xfT + 512*32;         // [bb*32+o][col]
  _Float16* Cfp = of + 512*32;          // packed fwd-DFT B tiles (ks*2+nt)
  _Float16* Cip = Cfp + 8*512;          // packed inv-DFT B tiles (nt)
  const int tid = (int)threadIdx.x;
  const int wv  = tid >> 5;
  const int b0  = blockIdx.x * 16;

  // generate packed DFT matrices directly in lane-swizzled B layout
  for (int t = tid; t < 8*512; t += 256) {       // Cfp: K=l-pos, N=2m|2m+1
    int j = t & 15, ln = (t >> 4) & 31, nt = (t >> 9) & 1, ks = t >> 10;
    int K = ks*32 + kof(ln, j);
    int N = nt*16 + (ln & 15);
    int md = N >> 1, ri = N & 1;
    float th = 6.2831853071795864f * (float)(md * K) / (float)SEQL;
    Cfp[t] = (_Float16)(ri ? -sinf(th) : cosf(th));
  }
  for (int t = tid; t < 8*512; t += 256) {       // Cip: K=2m|2m+1, N=l-pos
    int j = t & 15, ln = (t >> 4) & 31, nt = t >> 9;
    int K = kof(ln, j);
    int N = nt*16 + (ln & 15);
    int md = K >> 1, ri = K & 1;
    float sc = (md == 0) ? (1.0f/SEQL) : (2.0f/SEQL);
    float th = 6.2831853071795864f * (float)(md * N) / (float)SEQL;
    float v = ri ? ((md == 0) ? 0.f : (-sc * sinf(th))) : (sc * cosf(th));
    Cip[t] = (_Float16)v;
  }
  // fc0 lift (vectorized 8-wide stores)
  for (int t = tid; t < 16*WCH*(SEQL/8); t += 256) {
    int lc = (t & 15) * 8;
    int c  = (t >> 4) & 31;
    int bb = t >> 9;
    int gi = (b0 + bb)*SEQL + lc;
    float w0 = a.fc0w[c*3+0], w1 = a.fc0w[c*3+1], w2 = a.fc0w[c*3+2], wb = a.fc0b[c];
    v8h o8;
#pragma unroll
    for (int k = 0; k < 8; ++k)
      o8[k] = (_Float16)(w0*a.Y[gi+k] + w1*a.nn[gi+k] + w2*a.ed[gi+k] + wb);
    *(v8h*)(xT + (bb*WCH + c)*SEQL + lc) = o8;
  }
  __syncthreads();

  for (int lay = 0; lay < NLAY; ++lay) {
    const _Float16* swrp = a.swrp + (size_t)lay*NMODES*2*512;
    const _Float16* swip = a.swip + (size_t)lay*NMODES*2*512;
    const _Float16* pw   = a.pw16 + (size_t)lay*WCH*WCH;
    const float* pwb = a.pwb[lay];

    // ---- forward DFT: xfT[col][row] = (x[512,128] @ Cf[128,32])^T
    for (int job = wv; job < 64; job += 8) {
      int mt = job >> 1, nt = job & 1;
      v8f acc = {};
#pragma unroll
      for (int ks = 0; ks < 4; ++ks) {
        v16h A  = frag_a_c(xT + mt*16*SEQL + ks*32, SEQL);
        v16h Bf = frag_b_p(Cfp + (ks*2 + nt)*512);
        acc = wmma_f16(A, Bf, acc);
      }
      frag_store_h_mc(xfT + nt*16*512 + mt*16, 512, acc);   // lane-contiguous
    }
    __syncthreads();

    // ---- spectral mix per mode: of[b,o] = xf[b,i] * (wr + i wi)
    for (int job = wv; job < 2*NMODES; job += 8) {
      int md = job >> 1, nt = job & 1;
      v16h Ar = frag_a_c(xfT + (2*md)*512, 32);
      v16h Ai = frag_a_c(xfT + (2*md + 1)*512, 32);
      v16h An = -Ai;
      v16h Br = frag_b_p(swrp + ((size_t)md*2 + nt)*512);
      v16h Bi = frag_b_p(swip + ((size_t)md*2 + nt)*512);
      v8f cr = {}; cr = wmma_f16(Ar, Br, cr); cr = wmma_f16(An, Bi, cr);
      v8f ci = {}; ci = wmma_f16(Ar, Bi, ci); ci = wmma_f16(Ai, Br, ci);
      frag_store_h_g(of + nt*16*32 + 2*md,     1024, 32, cr);
      frag_store_h_g(of + nt*16*32 + 2*md + 1, 1024, 32, ci);
    }
    __syncthreads();

    // ---- inverse DFT + pointwise conv + gelu, per batch row (in-place on x)
    for (int bb = 0; bb < 16; ++bb) {
      v8f acc[2];
#pragma unroll
      for (int s = 0; s < 2; ++s) {
        int job = wv + s*8;
        int mt = job >> 3, nt = job & 7;
        v8f c = {};
        v16h Ao = frag_a_c(of + (bb*WCH + mt*16)*32, 32);
        v16h Bc = frag_b_p(Cip + nt*512);
        c = wmma_f16(Ao, Bc, c);
        v16h Ap = frag_a_c(pw + mt*16*WCH, WCH);
        v16h Bx = frag_b_g(xT + bb*WCH*SEQL + nt*16, SEQL, 1);  // dynamic gather
        c = wmma_f16(Ap, Bx, c);
        acc[s] = c;
      }
      __syncthreads();
#pragma unroll
      for (int s = 0; s < 2; ++s) {
        int job = wv + s*8;
        int mt = job >> 3, nt = job & 7;
        int l = lane_id(); int n = l & 15, mb = (l < 16) ? 0 : 8;
#pragma unroll
        for (int r = 0; r < 8; ++r) {
          int o = mt*16 + mb + r;
          xT[(bb*WCH + o)*SEQL + nt*16 + n] = (_Float16)gelu_tanh(acc[s][r] + pwb[o]);
        }
      }
      __syncthreads();
    }
  }

  // ---- global pooling over L -> g[16][32] (reuse xfT rows 0..511)
  for (int t = tid; t < 16*WCH; t += 256) {
    float s = 0.f;
    const v8h* row = (const v8h*)(xT + t*SEQL);
#pragma unroll 4
    for (int l8 = 0; l8 < SEQL/8; ++l8) {
      v8h v = row[l8];
#pragma unroll
      for (int k = 0; k < 8; ++k) s += (float)v[k];
    }
    xfT[t] = (_Float16)(s * (1.0f/SEQL));
  }
  __syncthreads();
  // ---- fc1: [16,32]@[32,64], gelu -> h (reuse `of` rows)
  if (wv < 4) {
    int nt = wv;
    v16h A  = frag_a_c(xfT, 32);
    v16h Bf = frag_b_p(a.fc1p + nt*512);
    v8f c = {}; c = wmma_f16(A, Bf, c);
    int l = lane_id(); int n = l & 15, mb = (l < 16) ? 0 : 8;
#pragma unroll
    for (int r = 0; r < 8; ++r) {
      int h = nt*16 + n;
      of[(mb + r)*WFD + h] = (_Float16)gelu_tanh(c[r] + a.fc1b[h]);
    }
  }
  __syncthreads();
  // ---- fc2: [16,64]@[64,32] -> mf2
  if (wv < 2) {
    int nt = wv;
    v8f c = {};
#pragma unroll
    for (int ks = 0; ks < 2; ++ks) {
      v16h A  = frag_a_c(of + ks*32, WFD);
      v16h Bf = frag_b_p(a.fc2p + (ks*2 + nt)*512);
      c = wmma_f16(A, Bf, c);
    }
    int l = lane_id(); int n = l & 15, mb = (l < 16) ? 0 : 8;
#pragma unroll
    for (int r = 0; r < 8; ++r) {
      int o = nt*16 + n;
      a.mf2[(size_t)(b0 + mb + r)*ODIM + o] = c[r] + a.fc2b[o];
    }
  }
}

// ==================== kernel 2: PICNN q-independent precompute ====================
struct PreArgs {
  const float* mf2;
  const float *Wu, *bu, *Wzu1, *bzu1, *Wuz1, *bz1, *Wzu2, *bzu2, *Wuz0, *bz0;
  const float *Wyu0, *byu0, *Wyu1, *byu1, *Wyu2, *byu2;
  float *c0, *c1, *g1, *g2, *a0, *y1, *y2;
};

__global__ void __launch_bounds__(64) picnn_pre_kernel(PreArgs a) {
  __shared__ float sx[ODIM], su[ZDIM];
  const int b = blockIdx.x, z = (int)threadIdx.x;
  if (z < ODIM) sx[z] = a.mf2[(size_t)b*ODIM + z];
  __syncthreads();
  float acc = a.bu[z];
  for (int c = 0; c < ODIM; ++c) acc += sx[c] * a.Wu[z*ODIM + c];
  su[z] = fmaxf(acc, 0.f);
  __syncthreads();
  float g1v = a.bzu1[z], c1v = a.bz1[z], g2v = a.bzu2[z];
  for (int j = 0; j < ZDIM; ++j) {
    float u = su[j];
    g1v += u * a.Wzu1[z*ZDIM + j];
    c1v += u * a.Wuz1[z*ZDIM + j];
    g2v += u * a.Wzu2[z*ZDIM + j];
  }
  float c0v = a.bz0[z];
  for (int c = 0; c < ODIM; ++c) c0v += sx[c] * a.Wuz0[z*ODIM + c];
  size_t idx = (size_t)b*ZDIM + z;
  a.g1[idx] = fmaxf(g1v, 0.f);
  a.c1[idx] = c1v;
  a.g2[idx] = fmaxf(g2v, 0.f);
  a.c0[idx] = c0v;
  if (z == 0) {
    float a0v = a.byu0[0]; for (int c = 0; c < ODIM; ++c) a0v += sx[c]*a.Wyu0[c];
    float y1v = a.byu1[0]; for (int j = 0; j < ZDIM; ++j) y1v += su[j]*a.Wyu1[j];
    float y2v = a.byu2[0]; for (int j = 0; j < ZDIM; ++j) y2v += su[j]*a.Wyu2[j];
    a.a0[b] = a0v; a.y1[b] = y1v; a.y2[b] = y2v;
  }
}

// ==================== kernel 3: viscoplastic scan (WMMA per timestep) ====================
struct ScanArgs {
  const float* e;
  const float *Wy0, *Wy1, *Wy2, *Wz2;
  const float *c0, *c1, *g1, *g2, *a0, *y1, *y2, *m1;
  const _Float16* awz1p;
  float* out;
};

__global__ void __launch_bounds__(256) scan_kernel(ScanArgs a) {
  __shared__ float s_c0[16*ZDIM], s_c1[16*ZDIM], s_g1[16*ZDIM], s_g2w[16*ZDIM];
  __shared__ alignas(32) _Float16 s_W[ZDIM*ZDIM];   // packed |Wz1|^T B tiles
  __shared__ alignas(32) _Float16 s_A[32*ZDIM];     // [b | 16+b][z] activations (f16)
  __shared__ float s_D[32*ZDIM];
  __shared__ float s_wy0[ZDIM], s_wy1[ZDIM];
  __shared__ float s_q[16], s_a0[16], s_y1[16], s_gy2[16], s_m1[16], s_xi[16];
  const int tid = (int)threadIdx.x, wv = tid >> 5;
  const int b0 = blockIdx.x * 16;

#if VP_HAVE_TDM
  // Tensor Data Mover: DMA the 8KB packed |Wz1|^T straight into LDS.
  if (wv == 0) {
    unsigned lds_off = (unsigned)(size_t)(&s_W[0]);
    unsigned long long ga = (unsigned long long)(size_t)a.awz1p;
    u32x4_t g0 = { 1u,                                   // count=1 (valid user D#)
                   lds_off,                              // lds_addr
                   (unsigned)(ga & 0xFFFFFFFFu),         // global_addr[31:0]
                   (unsigned)((ga >> 32) & 0x01FFFFFFu) | (2u << 30) }; // addr hi | type=2
    i32x8_t g1 = { (int)0x00010000,                      // data_size=1 -> 2 bytes
                   (int)((4096u & 0xFFFFu) << 16),       // tensor_dim0=4096 (lo16)
                   (int)(1u << 16),                      // tensor_dim0 hi=0 | tensor_dim1=1
                   (int)((4096u & 0xFFFFu) << 16),       // tensor_dim1 hi=0 | tile_dim0=4096
                   (int)1,                               // tile_dim1=1 | tile_dim2=0
                   (int)4096,                            // tensor_dim0_stride lo32
                   0, 0 };                               // stride hi / dim1_stride
    i32x4_t gz = { 0, 0, 0, 0 };
    i32x8_t gz8 = { 0, 0, 0, 0, 0, 0, 0, 0 };
    __builtin_amdgcn_tensor_load_to_lds(g0, g1, gz, gz, gz8, 0);
    __builtin_amdgcn_s_wait_tensorcnt(0);
  }
#else
  for (int t = tid; t < ZDIM*ZDIM; t += 256) s_W[t] = a.awz1p[t];
#endif

  for (int t = tid; t < 16*ZDIM; t += 256) {
    int b = t >> 6, z = t & 63;
    size_t gi = (size_t)(b0 + b)*ZDIM + z;
    s_c0[t] = a.c0[gi]; s_c1[t] = a.c1[gi]; s_g1[t] = a.g1[gi];
    s_g2w[t] = a.g2[gi] * fabsf(a.Wz2[z]);
  }
  if (tid < ZDIM) { s_wy0[tid] = a.Wy0[tid]; s_wy1[tid] = a.Wy1[tid]; }
  if (tid < 16) {
    int b = b0 + tid;
    s_a0[tid]  = a.a0[b];
    s_y1[tid]  = a.y1[b];
    s_gy2[tid] = a.y2[b] * a.Wy2[0];
    s_m1[tid]  = a.m1[b];
    s_xi[tid]  = 0.f;
  }
  __syncthreads();

  for (int ts = 0; ts < TSTEPS; ++ts) {
    if (tid < 16) {   // q = m1*(e - xi); emit stress & xi (pre-update, like scan)
      int b = tid;
      float q = s_m1[b] * (a.e[(size_t)(b0 + b)*TSTEPS + ts] - s_xi[b]);
      s_q[b] = q;
      a.out[(size_t)(b0 + b)*TSTEPS + ts] = q;
      a.out[(size_t)BATCH*TSTEPS + (size_t)(b0 + b)*TSTEPS + ts] = s_xi[b];
    }
    __syncthreads();
    // z1 = sqrelu(q*d0 + c0); rows: [b] -> z1*g1 ; [16+b] -> (dz1/dq)*g1
    for (int t = tid; t < 16*ZDIM; t += 256) {
      int b = t >> 6, z = t & 63;
      float d0  = s_a0[b] * s_wy0[z];
      float pre = s_q[b]*d0 + s_c0[t];
      float r   = fmaxf(pre, 0.f);
      float g   = s_g1[t];
      s_A[t]            = (_Float16)(r*r*g);
      s_A[16*ZDIM + t]  = (_Float16)(2.f*r*d0*g);
    }
    __syncthreads();
    {   // [32,64] @ |Wz1|^T[64,64]  -> 8 waves, one 16x16 tile each, K=2 steps
      int mt = wv >> 2, nt = wv & 3;
      v8f c = {};
#pragma unroll
      for (int ks = 0; ks < 2; ++ks) {
        v16h A  = frag_a_c(s_A + mt*16*ZDIM + ks*32, ZDIM);
        v16h Bf = frag_b_p(s_W + (ks*4 + nt)*512);
        c = wmma_f16(A, Bf, c);
      }
      frag_store_f(s_D + mt*16*ZDIM + nt*16, ZDIM, 1, c);
    }
    __syncthreads();
    {   // dout/dq = (2 relu(z2pre) * dz2pre/dq * g2) . |Wz2| + y2*Wy2
      int b = tid >> 4, sl = tid & 15;
      float part = 0.f;
#pragma unroll
      for (int k = 0; k < 4; ++k) {
        int z = sl + 16*k;
        float yw  = s_y1[b] * s_wy1[z];
        float z2p = s_D[b*ZDIM + z] + s_q[b]*yw + s_c1[b*ZDIM + z];
        float u   = s_D[(16 + b)*ZDIM + z] + yw;
        part += 2.f * fmaxf(z2p, 0.f) * u * s_g2w[b*ZDIM + z];
      }
      part += __shfl_xor(part, 1);
      part += __shfl_xor(part, 2);
      part += __shfl_xor(part, 4);
      part += __shfl_xor(part, 8);
      if (sl == 0) s_xi[b] += DTC * (part + s_gy2[b]);
    }
    __syncthreads();
  }
}

} // namespace vp

extern "C" void kernel_launch(void* const* d_in, const int* in_sizes, int n_in,
                              void* d_out, int out_size, void* d_ws, size_t ws_size,
                              hipStream_t stream) {
  using namespace vp;
  (void)in_sizes; (void)n_in; (void)out_size; (void)ws_size;
  auto F = [&](int i) { return (const float*)d_in[i]; };
  char* ws = (char*)d_ws;

  float* m1  = (float*)(ws + WS_M1);
  float* mf2 = (float*)(ws + WS_MF2);
  float* c0  = (float*)(ws + WS_C0);
  float* c1  = (float*)(ws + WS_C1);
  float* g1  = (float*)(ws + WS_G1);
  float* g2  = (float*)(ws + WS_G2);
  float* a0  = (float*)(ws + WS_A0);
  float* y1  = (float*)(ws + WS_Y1);
  float* y2  = (float*)(ws + WS_Y2);
  _Float16* swrp  = (_Float16*)(ws + WS_SWRP);
  _Float16* swip  = (_Float16*)(ws + WS_SWIP);
  _Float16* pw16  = (_Float16*)(ws + WS_PW);
  _Float16* fc1p  = (_Float16*)(ws + WS_FC1P);
  _Float16* fc2p  = (_Float16*)(ws + WS_FC2P);
  _Float16* awz1p = (_Float16*)(ws + WS_AWZ1P);

  // ---- kernel 0: prep ----
  PrepArgs pa;
  pa.E = F(IN_E);
  pa.swr[0] = F(IN_sw0_r); pa.swr[1] = F(IN_sw1_r); pa.swr[2] = F(IN_sw2_r);
  pa.swi[0] = F(IN_sw0_i); pa.swi[1] = F(IN_sw1_i); pa.swi[2] = F(IN_sw2_i);
  pa.pww[0] = F(IN_pw0_w); pa.pww[1] = F(IN_pw1_w); pa.pww[2] = F(IN_pw2_w);
  pa.fc1w = F(IN_fc1_w); pa.fc2w = F(IN_fc2_w); pa.Wz1 = F(IN_Wz1);
  pa.m1 = m1; pa.swrp = swrp; pa.swip = swip; pa.pw16 = pw16;
  pa.fc1p = fc1p; pa.fc2p = fc2p; pa.awz1p = awz1p;
  const int prepTasks = BATCH + NLAY*NMODES*2*512 + NLAY*WCH*WCH
                      + 4*512 + 4*512 + 8*512;
  prep_kernel<<<(prepTasks + 255)/256, 256, 0, stream>>>(pa);

  // ---- kernel 1: FNO encoder ----
  FnoArgs fa;
  fa.Y = F(IN_Y); fa.nn = F(IN_n); fa.ed = F(IN_edot);
  fa.fc0w = F(IN_fc0_w); fa.fc0b = F(IN_fc0_b);
  fa.fc1b = F(IN_fc1_b); fa.fc2b = F(IN_fc2_b);
  fa.pwb[0] = F(IN_pw0_b); fa.pwb[1] = F(IN_pw1_b); fa.pwb[2] = F(IN_pw2_b);
  fa.swrp = swrp; fa.swip = swip; fa.pw16 = pw16;
  fa.fc1p = fc1p; fa.fc2p = fc2p; fa.mf2 = mf2;
  fno_kernel<<<BATCH/16, 256, (unsigned)FNO_LDS_BYTES, stream>>>(fa);

  // ---- kernel 2: PICNN precompute ----
  PreArgs pr;
  pr.mf2 = mf2;
  pr.Wu = F(IN_Wu); pr.bu = F(IN_bu);
  pr.Wzu1 = F(IN_Wzu1); pr.bzu1 = F(IN_bzu1);
  pr.Wuz1 = F(IN_Wuz1); pr.bz1 = F(IN_bz1);
  pr.Wzu2 = F(IN_Wzu2); pr.bzu2 = F(IN_bzu2);
  pr.Wuz0 = F(IN_Wuz0); pr.bz0 = F(IN_bz0);
  pr.Wyu0 = F(IN_Wyu0); pr.byu0 = F(IN_byu0);
  pr.Wyu1 = F(IN_Wyu1); pr.byu1 = F(IN_byu1);
  pr.Wyu2 = F(IN_Wyu2); pr.byu2 = F(IN_byu2);
  pr.c0 = c0; pr.c1 = c1; pr.g1 = g1; pr.g2 = g2;
  pr.a0 = a0; pr.y1 = y1; pr.y2 = y2;
  picnn_pre_kernel<<<BATCH, 64, 0, stream>>>(pr);

  // ---- kernel 3: time-integration scan ----
  ScanArgs sa;
  sa.e = F(IN_e);
  sa.Wy0 = F(IN_Wy0); sa.Wy1 = F(IN_Wy1); sa.Wy2 = F(IN_Wy2); sa.Wz2 = F(IN_Wz2);
  sa.c0 = c0; sa.c1 = c1; sa.g1 = g1; sa.g2 = g2;
  sa.a0 = a0; sa.y1 = y1; sa.y2 = y2; sa.m1 = m1;
  sa.awz1p = awz1p;
  sa.out = (float*)d_out;
  scan_kernel<<<BATCH/16, 256, 0, stream>>>(sa);
}
